// SingleHeadedAttention_5772436046421
// MI455X (gfx1250) — matrix operand order repaired and checked by reading, not compiled, not run on hardware
//
#include <hip/hip_runtime.h>
#include <hip/hip_fp16.h>

typedef __attribute__((ext_vector_type(16))) _Float16 v16h;
typedef __attribute__((ext_vector_type(8)))  _Float16 v8h;
typedef __attribute__((ext_vector_type(8)))  float    v8f;

#define BATCH 4
#define SEQ   2048
#define DIM   1024
#define DQK   128
#define DV    1024
#define ROWS  (BATCH * SEQ)   // 8192

// ---------------------------------------------------------------------------
// Fragment loaders (contiguous k order k = 16h + s, same for A and B; wave32)
// A 16x32 f16: lane r=lane&15 holds row r; h=lane>>4 selects the contiguous
// columns [h*16 .. h*16+15].
// ---------------------------------------------------------------------------
__device__ __forceinline__ v16h load_a_frag(const _Float16* base, int stride, int lane) {
    const int r = lane & 15, h = lane >> 4;
    const _Float16* p = base + (size_t)r * stride + h * 16;
    v8h lo = *(const v8h*)(p);
    v8h hi = *(const v8h*)(p + 8);
    v16h out;
#pragma unroll
    for (int i = 0; i < 8; ++i) { out[i] = lo[i]; out[i + 8] = hi[i]; }
    return out;
}

// B 32x16 f16: lane holds column n=lane&15, K-range (lane>>4)*16 .. +15.
// `base` points at a matrix stored with the *output-column* index as the row
// (i.e. pre-transposed), so the 16 K-values are contiguous: one 32B load.
__device__ __forceinline__ v16h load_b_frag(const _Float16* base, int stride, int lane) {
    const int n = lane & 15, kk = lane >> 4;
    return *(const v16h*)(base + (size_t)n * stride + kk * 16);
}

// silu via v_rcp_f32 (avoids the IEEE v_div_scale/div_fmas chain)
__device__ __forceinline__ float silu_f(float x) {
    return x * __builtin_amdgcn_rcpf(1.0f + __expf(-x));
}

// laplacian_attn_fn: 0.5*(1+erf((x - sqrt(0.5)) / (sqrt(0.25*pi)*sqrt(2))))
__device__ __forceinline__ float laplacian_f(float x) {
    return 0.5f * (1.0f + erff((x - 0.70710678118654752f) * 0.79788456080286536f));
}

// ---------------------------------------------------------------------------
// Stage 0: conversions
// ---------------------------------------------------------------------------
__global__ __launch_bounds__(256) void cvt_f32_to_f16(const float* __restrict__ src,
                                                      _Float16* __restrict__ dst, int n) {
    for (int i = blockIdx.x * blockDim.x + threadIdx.x; i < n; i += gridDim.x * blockDim.x)
        dst[i] = (_Float16)src[i];
}

// w[K][N] -> wT[N][K] (f16)
__global__ __launch_bounds__(256) void cvt_transpose_f16(const float* __restrict__ w,
                                                         _Float16* __restrict__ wT,
                                                         int K, int N) {
    int total = K * N;
    for (int i = blockIdx.x * blockDim.x + threadIdx.x; i < total; i += gridDim.x * blockDim.x) {
        int k = i / N, n = i - k * N;
        wT[(size_t)n * K + k] = (_Float16)w[i];
    }
}

// Bias per distance d = i - j (0..SEQ-1): T5 bucketing, scaled by sqrt(DQK).
__global__ __launch_bounds__(256) void bias_tab_kernel(const float* __restrict__ rel,
                                                       float* __restrict__ tab) {
    int d = blockIdx.x * blockDim.x + threadIdx.x;
    if (d < SEQ) {
        int bucket;
        if (d < 16) {
            bucket = d;
        } else {
            int v = 16 + (int)(__logf((float)d * (1.0f / 16.0f)) * (16.0f / __logf(8.0f)));
            bucket = v < 31 ? v : 31;
        }
        tab[d] = rel[bucket] * 11.31370849898476f;   // sqrt(128)
    }
}

// ---------------------------------------------------------------------------
// Stage 1: qk = silu(x @ wqk + bqk); q = qk*g0+b0, k = qk*g1+b1   (f16 out)
// grid = ROWS/16 blocks, 256 thr (8 waves cover N=128)
// ---------------------------------------------------------------------------
__global__ __launch_bounds__(256) void proj_qk_kernel(
    const _Float16* __restrict__ xh, const _Float16* __restrict__ wqkT,
    const float* __restrict__ bqk, const float* __restrict__ gamma,
    const float* __restrict__ beta,
    _Float16* __restrict__ qh, _Float16* __restrict__ kh) {
    const int lane = threadIdx.x & 31;
    const int wave = threadIdx.x >> 5;
    const int m0 = blockIdx.x * 16;
    const int n0 = wave * 16;

    v8f acc = {};
#pragma unroll 4
    for (int k0 = 0; k0 < DIM; k0 += 32) {
        v16h a = load_a_frag(xh + (size_t)m0 * DIM + k0, DIM, lane);
        v16h b = load_b_frag(wqkT + (size_t)n0 * DIM + k0, DIM, lane);
        acc = __builtin_amdgcn_wmma_f32_16x16x32_f16(false, a, false, b, (short)0, acc, false, false);
    }

    const int n = n0 + (lane & 15);
    const float bias = bqk[n];
    const float g0 = gamma[n], b0 = beta[n];
    const float g1 = gamma[DQK + n], b1 = beta[DQK + n];
    const int mbase = m0 + (lane >> 4) * 8;
#pragma unroll
    for (int e = 0; e < 8; ++e) {
        float act = silu_f(acc[e] + bias);
        size_t idx = (size_t)(mbase + e) * DQK + n;
        qh[idx] = (_Float16)(act * g0 + b0);
        kh[idx] = (_Float16)(act * g1 + b1);
    }
}

// ---------------------------------------------------------------------------
// Stage 2: v = silu(x @ wv + bv), stored transposed: vt[col][row] (f16)
// grid = (ROWS/16)*2 blocks; 8 waves x 64 cols = 512 cols per block
// ---------------------------------------------------------------------------
__global__ __launch_bounds__(256) void proj_v_kernel(
    const _Float16* __restrict__ xh, const _Float16* __restrict__ wvT,
    const float* __restrict__ bv, _Float16* __restrict__ vt) {
    const int lane = threadIdx.x & 31;
    const int wave = threadIdx.x >> 5;
    const int m0 = (blockIdx.x >> 1) * 16;
    const int n0 = (blockIdx.x & 1) * 512 + wave * 64;

    v8f acc[4] = {{}, {}, {}, {}};
#pragma unroll 2
    for (int k0 = 0; k0 < DIM; k0 += 32) {
        v16h a = load_a_frag(xh + (size_t)m0 * DIM + k0, DIM, lane);
#pragma unroll
        for (int t = 0; t < 4; ++t) {
            v16h b = load_b_frag(wvT + (size_t)(n0 + t * 16) * DIM + k0, DIM, lane);
            acc[t] = __builtin_amdgcn_wmma_f32_16x16x32_f16(false, a, false, b, (short)0, acc[t], false, false);
        }
    }

    const int mbase = m0 + (lane >> 4) * 8;
#pragma unroll
    for (int t = 0; t < 4; ++t) {
        const int n = n0 + t * 16 + (lane & 15);
        const float bias = bv[n];
        v8h pk;
#pragma unroll
        for (int e = 0; e < 8; ++e) pk[e] = (_Float16)silu_f(acc[t][e] + bias);
        // C-frag rows are contiguous along vt's inner (row) dimension
        *(v8h*)(vt + (size_t)n * ROWS + mbase) = pk;
    }
}

// ---------------------------------------------------------------------------
// Stage 3: flash-style Laplacian attention (no softmax state needed).
// One block owns 16 query rows x ALL 1024 v-cols (8 waves x 128 cols).
// Per super-iteration of 256 keys: wave w produces the P tile (16x32 f16) for
// key block J+32w into LDS (S = Q.K^T, 8 WMMAs, + scale/bias/erf/causal mask,
// computed exactly once per element), then after a barrier every wave consumes
// all active P tiles against its 128-col slice of V (8 C-frags = 64 VGPRs).
// grid = BATCH * (SEQ/16) blocks, 256 thr.
// ---------------------------------------------------------------------------
__global__ __launch_bounds__(256) void attn_kernel(
    const _Float16* __restrict__ qh, const _Float16* __restrict__ kh,
    const _Float16* __restrict__ vt, const float* __restrict__ biasTab,
    float* __restrict__ out) {
    const int lane = threadIdx.x & 31;
    const int wave = threadIdx.x >> 5;

    int task = blockIdx.x;
    const int qt = task & 127;
    const int bb = task >> 7;             // batch
    const int i0 = qt * 16;               // query tile start (in-sequence)
    const int n0 = wave * 128;            // this wave's v-column slice

    __shared__ _Float16 Pl[8][16 * 32];   // 8 key-block P tiles, 8 KB

    const size_t rowq = (size_t)bb * SEQ + i0;

    // Persistent Q fragments: 16 x 128 as 4 A-frags
    v16h qa[4];
#pragma unroll
    for (int c = 0; c < 4; ++c)
        qa[c] = load_a_frag(qh + rowq * DQK + c * 32, DQK, lane);

    v8f o[8] = {{}, {}, {}, {}, {}, {}, {}, {}};

    const float scale = 1.0f / (float)SEQ;
    const int nl = lane & 15;
    const int mb = (lane >> 4) * 8;
    const int jmax = i0 + 15;             // last valid key for this tile

    for (int J = 0; J <= jmax; J += 256) {
        int nb = (jmax - J) / 32 + 1;     // active 32-key blocks this round
        if (nb > 8) nb = 8;

        // ---- producer: this wave's key block (if active)
        if (wave < nb) {
            const int j0 = J + wave * 32;
#pragma unroll
            for (int js = 0; js < 2; ++js) {
                const size_t rowk = (size_t)bb * SEQ + j0 + js * 16;
                v8f s = {};
#pragma unroll
                for (int c = 0; c < 4; ++c) {
                    v16h kb = load_b_frag(kh + rowk * DQK + c * 32, DQK, lane);
                    s = __builtin_amdgcn_wmma_f32_16x16x32_f16(false, qa[c], false, kb, (short)0, s, false, false);
                }
                const int j = j0 + js * 16 + nl;   // key index
#pragma unroll
                for (int e = 0; e < 8; ++e) {
                    const int i = i0 + mb + e;     // query index
                    float p = 0.0f;
                    if (j <= i)
                        p = laplacian_f(s[e] * scale + biasTab[i - j]);
                    Pl[wave][(size_t)(mb + e) * 32 + js * 16 + nl] = (_Float16)p;
                }
            }
        }
        __syncthreads();

        // ---- consumer: O += P . V over all active key blocks
        for (int w = 0; w < nb; ++w) {
            const int j0 = J + w * 32;
            v16h pa = load_a_frag(&Pl[w][0], 32, lane);
#pragma unroll
            for (int t = 0; t < 8; ++t) {
                v16h vb = load_b_frag(vt + (size_t)(n0 + t * 16) * ROWS + bb * SEQ + j0, ROWS, lane);
                o[t] = __builtin_amdgcn_wmma_f32_16x16x32_f16(false, pa, false, vb, (short)0, o[t], false, false);
            }
        }
        __syncthreads();                  // P tiles consumed before next round
    }

    // ---- store output (f32)
#pragma unroll
    for (int t = 0; t < 8; ++t) {
        const int col = n0 + t * 16 + nl;
#pragma unroll
        for (int e = 0; e < 8; ++e) {
            const int i = i0 + mb + e;
            out[((size_t)bb * SEQ + i) * DV + col] = o[t][e];
        }
    }
}

// ---------------------------------------------------------------------------
// Host-side launcher
// ---------------------------------------------------------------------------
extern "C" void kernel_launch(void* const* d_in, const int* in_sizes, int n_in,
                              void* d_out, int out_size, void* d_ws, size_t ws_size,
                              hipStream_t stream) {
    const float* x     = (const float*)d_in[0];   // [4,2048,1024]
    const float* wqk   = (const float*)d_in[1];   // [1024,128]
    const float* bqk   = (const float*)d_in[2];   // [128]
    const float* gamma = (const float*)d_in[3];   // [2,128]
    const float* beta  = (const float*)d_in[4];   // [2,128]
    const float* wv    = (const float*)d_in[5];   // [1024,1024]
    const float* bv    = (const float*)d_in[6];   // [1024]
    const float* rel   = (const float*)d_in[7];   // [32]
    float* out = (float*)d_out;

    char* ws = (char*)d_ws;
    size_t off = 0;
    auto carve = [&](size_t bytes) { char* p = ws + off; off += (bytes + 255) & ~(size_t)255; return p; };
    _Float16* xh    = (_Float16*)carve((size_t)ROWS * DIM * 2);   // 16 MB
    _Float16* wqkT  = (_Float16*)carve((size_t)DQK * DIM * 2);    // 256 KB
    _Float16* wvT   = (_Float16*)carve((size_t)DV * DIM * 2);     // 2 MB
    _Float16* qh    = (_Float16*)carve((size_t)ROWS * DQK * 2);   // 2 MB
    _Float16* kh    = (_Float16*)carve((size_t)ROWS * DQK * 2);   // 2 MB
    _Float16* vt    = (_Float16*)carve((size_t)DV * ROWS * 2);    // 16 MB
    float*    btab  = (float*)   carve((size_t)SEQ * 4);          // 8 KB

    // Stage 0: conversions + bias table
    cvt_f32_to_f16<<<4096, 256, 0, stream>>>(x, xh, ROWS * DIM);
    cvt_transpose_f16<<<512, 256, 0, stream>>>(wqk, wqkT, DIM, DQK);
    cvt_transpose_f16<<<4096, 256, 0, stream>>>(wv, wvT, DIM, DV);
    bias_tab_kernel<<<SEQ / 256, 256, 0, stream>>>(rel, btab);

    // Stage 1/2: projections (WMMA GEMMs)
    proj_qk_kernel<<<ROWS / 16, 256, 0, stream>>>(xh, wqkT, bqk, gamma, beta, qh, kh);
    proj_v_kernel<<<(ROWS / 16) * 2, 256, 0, stream>>>(xh, wvT, bv, vt);

    // Stage 3: attention (one block per 16-query stripe, all columns)
    attn_kernel<<<BATCH * (SEQ / 16), 256, 0, stream>>>(qh, kh, vt, btab, out);
}
